// MambaBlock_19112604467503
// MI455X (gfx1250) — matrix operand order repaired and checked by reading, not compiled, not run on hardware
//
#include <hip/hip_runtime.h>

// ---------------- problem constants ----------------
#define B_   2
#define L_   1024
#define DM_  1024
#define DI_  2048
#define NS_  16     // d_state
#define RK_  64     // dt_rank
#define XPD_ 96     // dt_rank + 2*d_state
#define M_   (B_ * L_)   // 2048 rows for all GEMMs

// ---------------- types ----------------
typedef __bf16 bf16_t;
typedef __attribute__((ext_vector_type(16))) __bf16        v16bf;
typedef __attribute__((ext_vector_type(8)))  float         v8f;
typedef __attribute__((ext_vector_type(4)))  unsigned int  uint4v;

union FragBF {
    v16bf  bf;
    uint4v u4[2];
};

__device__ __forceinline__ bf16_t f2bf(float f) {
    unsigned int u = __float_as_uint(f);
    u += 0x7fffu + ((u >> 16) & 1u);            // round to nearest even
    unsigned short h = (unsigned short)(u >> 16);
    bf16_t r;
    __builtin_memcpy(&r, &h, 2);
    return r;
}

// ---------------- f32 -> bf16 convert (dense) ----------------
__global__ __launch_bounds__(256) void cvt_bf16_kernel(const float* __restrict__ src,
                                                       bf16_t* __restrict__ dst, int n) {
    int i = blockIdx.x * 256 + threadIdx.x;
    if (i < n) dst[i] = f2bf(src[i]);
}

// ---------------- strided f32 -> bf16 (extract dt_low cols 0..63 of xp) ----------------
__global__ __launch_bounds__(256) void cvt_bf16_strided_kernel(const float* __restrict__ src,
                                                               bf16_t* __restrict__ dst, int n) {
    int i = blockIdx.x * 256 + threadIdx.x;
    if (i < n) {
        int row = i / RK_;
        int c   = i - row * RK_;
        dst[i] = f2bf(src[(size_t)row * XPD_ + c]);
    }
}

// ---------------- RMSNorm -> bf16 ----------------
__global__ __launch_bounds__(256) void rmsnorm_bf16_kernel(const float* __restrict__ u,
                                                           const float* __restrict__ w,
                                                           bf16_t* __restrict__ out) {
    __shared__ float red[8];
    const int row = blockIdx.x;
    const float* up = u + (size_t)row * DM_;
    float s = 0.f;
    for (int i = threadIdx.x; i < DM_; i += 256) { float v = up[i]; s += v * v; }
    #pragma unroll
    for (int off = 16; off >= 1; off >>= 1) s += __shfl_xor(s, off, 32);
    if ((threadIdx.x & 31) == 0) red[threadIdx.x >> 5] = s;
    __syncthreads();
    if (threadIdx.x < 8) {
        float t = red[threadIdx.x];
        #pragma unroll
        for (int off = 4; off >= 1; off >>= 1) t += __shfl_xor(t, off, 8);
        if (threadIdx.x == 0) red[0] = t;
    }
    __syncthreads();
    const float scale = rsqrtf(red[0] * (1.0f / DM_) + 1e-5f);
    for (int i = threadIdx.x; i < DM_; i += 256)
        out[(size_t)row * DM_ + i] = f2bf(up[i] * scale * w[i]);
}

// ---------------- generic bf16 WMMA GEMM: C[M,N] = A[M,K] * W[N,K]^T (+residual) --------
// Block = 128 threads = 4 waves sharing one (NT*16)-column B range; each wave owns a
// 16-row M slice. B is staged in 64-deep K tiles into LDS via the CDNA5 async-copy path
// (global_load_async_to_lds_b128 / ASYNCcnt), double-buffered so the DMA of tile k+64
// overlaps the 2*NT WMMAs of tile k (one barrier per 2*NT WMMAs). LDS rows padded to
// 144B (36 dwords) -> conflict-free b128 reads. All NT B fragments of a sub-step are
// loaded before the WMMA chain so s_wait_dscnt staggers instead of stalling per WMMA.
// Fragment layouts per CDNA5 ISA 7.12.2 (wave32). Grid must tile M,N exactly; K % 64 == 0.
template <int NT, bool RES>
__global__ __launch_bounds__(128) void gemm_bf16_wmma_kernel(
    const bf16_t* __restrict__ A, int lda,
    const bf16_t* __restrict__ W, int ldw,
    float* __restrict__ C, int ldc,
    const float* __restrict__ residual,
    int K)
{
    constexpr int ROWB  = 144;              // padded LDS row pitch in bytes (64 K * 2B + pad)
    constexpr int TILEB = NT * 16 * ROWB;   // bytes per K-tile buffer
    __shared__ unsigned char smem[2 * TILEB];

    const int lane = threadIdx.x & 31;
    const int wave = threadIdx.x >> 5;                 // 0..3 -> M sub-tile
    const int m0   = blockIdx.y * 64 + wave * 16;
    const int n0   = blockIdx.x * (NT * 16);

    const int half = lane >> 4;                        // K-half selector
    const int l16  = lane & 15;

    // flat shared address truncated to 32 bits == LDS byte offset (ISA 10.2 aperture map)
    const unsigned lds0 = (unsigned)(uintptr_t)(void*)smem;

    v8f acc[NT] = {};

    const bf16_t* ap = A + (size_t)(m0 + l16) * lda + half * 8;   // 16-bit A 16x32 layout

    // async-copy one (NT*16) x 64 B-tile (K-slice of W rows n0..) into an LDS buffer
    auto issue_tile = [&](int kk, unsigned ldsbase) {
        #pragma unroll
        for (int c = 0; c < NT; ++c) {
            const int chunk = threadIdx.x + c * 128;   // (NT*16 rows) * 8 x 16B segments
            const int row   = chunk >> 3;
            const int seg   = chunk & 7;
            const unsigned long long ga =
                (unsigned long long)(uintptr_t)(W + (size_t)(n0 + row) * ldw + kk + seg * 8);
            const unsigned loff = ldsbase + (unsigned)(row * ROWB + seg * 16);
            asm volatile("global_load_async_to_lds_b128 %0, %1, off"
                         :: "v"(loff), "v"(ga) : "memory");
        }
    };

    issue_tile(0, lds0);

    int buf = 0;
    for (int k = 0; k < K; k += 64, buf ^= 1) {
        // A fragments for both sub-steps, issued before the fence so latency hides under it
        FragBF a0, a1;
        a0.u4[0] = *(const uint4v*)(ap + k);
        a0.u4[1] = *(const uint4v*)(ap + k + 16);
        a1.u4[0] = *(const uint4v*)(ap + k + 32);
        a1.u4[1] = *(const uint4v*)(ap + k + 48);
        __builtin_prefetch(ap + k + 64, 0, 1);         // next A tile -> global_prefetch_b8

        asm volatile("s_wait_asynccnt 0x0" ::: "memory");  // my writes of buf done
        __syncthreads();                                   // everyone's writes of buf done
        if (k + 64 < K) issue_tile(k + 64, lds0 + (unsigned)((buf ^ 1) * TILEB));

        const unsigned char* sbuf = smem + buf * TILEB;
        #pragma unroll
        for (int s = 0; s < 2; ++s) {
            const FragBF& a = s ? a1 : a0;
            FragBF b[NT];                               // B 32x16 layout from LDS
            #pragma unroll
            for (int t = 0; t < NT; ++t) {
                const unsigned char* bp = sbuf + (t * 16 + l16) * ROWB + s * 64 + half * 16;
                b[t].u4[0] = *(const uint4v*)(bp);
                b[t].u4[1] = *(const uint4v*)(bp + 32);
            }
            #pragma unroll
            for (int t = 0; t < NT; ++t)
                acc[t] = __builtin_amdgcn_wmma_f32_16x16x32_bf16(
                    false, a.bf, false, b[t].bf, (short)0, acc[t], false, false);
        }
    }

    #pragma unroll
    for (int t = 0; t < NT; ++t) {
        const int col = n0 + t * 16 + l16;
        #pragma unroll
        for (int r = 0; r < 8; ++r) {
            const int row = m0 + half * 8 + r;         // f32 C/D 16x16 layout
            const size_t idx = (size_t)row * ldc + col;
            float v = acc[t][r];
            if (RES) v += residual[idx];
            C[idx] = v;
        }
    }
}

// ---------------- depthwise causal conv(4) + bias + SiLU ----------------
__global__ __launch_bounds__(256) void conv_silu_kernel(const float* __restrict__ xz,
                                                        const float* __restrict__ cw,
                                                        const float* __restrict__ cb,
                                                        float* __restrict__ xact,
                                                        bf16_t* __restrict__ xactb) {
    const int idx = blockIdx.x * 256 + threadIdx.x;    // over M_*DI_
    const int d  = idx & (DI_ - 1);
    const int bl = idx >> 11;                          // DI_ = 2048
    const int b  = bl >> 10;                           // L_  = 1024
    const int l  = bl & (L_ - 1);
    float acc = cb[d];
    #pragma unroll
    for (int k = 0; k < 4; ++k) {
        const int ll = l - 3 + k;
        if (ll >= 0)
            acc += cw[d * 4 + k] * xz[((size_t)(b * L_ + ll)) * (2 * DI_) + d];
    }
    const float s = acc / (1.f + __expf(-acc));
    xact[idx]  = s;
    xactb[idx] = f2bf(s);
}

// ---------------- dt = softplus(dt_pre + bias), in place ----------------
__global__ __launch_bounds__(256) void dt_softplus_kernel(float* __restrict__ dt,
                                                          const float* __restrict__ bias) {
    const int idx = blockIdx.x * 256 + threadIdx.x;    // over M_*DI_
    const int d = idx & (DI_ - 1);
    const float v = dt[idx] + bias[d];
    dt[idx] = (v > 20.f) ? v : log1pf(__expf(v));
}

// ---------------- selective scan + D skip + SiLU gating -> y_bf16 ----------------
// 16 lanes per (b,d) channel, one state element per lane, shfl_xor reduction for C dot.
__global__ __launch_bounds__(256) void scan_gate_kernel(const float* __restrict__ xact,
                                                        const float* __restrict__ dt,
                                                        const float* __restrict__ xp,
                                                        const float* __restrict__ xz,
                                                        const float* __restrict__ A_log,
                                                        const float* __restrict__ Dp,
                                                        bf16_t* __restrict__ ybf) {
    const int tid = blockIdx.x * 256 + threadIdx.x;    // over B_*DI_*NS_
    const int n = tid & (NS_ - 1);
    const int g = tid >> 4;
    const int b = g / DI_;
    const int d = g - b * DI_;
    const float Aa = -__expf(A_log[d * NS_ + n]);
    const float Dd = Dp[d];
    float state = 0.f;
    for (int t = 0; t < L_; ++t) {
        const size_t row = (size_t)(b * L_ + t);
        const float xv  = xact[row * DI_ + d];
        const float dtv = dt[row * DI_ + d];
        const float Bv  = xp[row * XPD_ + RK_ + n];
        const float Cv  = xp[row * XPD_ + RK_ + NS_ + n];
        const float dA  = __expf(Aa * dtv);
        state = dA * state + (dtv * xv) * Bv;
        float y = state * Cv;
        y += __shfl_xor(y, 1, 16);
        y += __shfl_xor(y, 2, 16);
        y += __shfl_xor(y, 4, 16);
        y += __shfl_xor(y, 8, 16);
        if (n == 0) {
            const float yy = y + xv * Dd;
            const float zv = xz[row * (2 * DI_) + DI_ + d];
            const float gate = zv / (1.f + __expf(-zv));
            ybf[row * DI_ + d] = f2bf(yy * gate);
        }
    }
}

// ---------------- host-side orchestration ----------------
static inline int cdiv(int a, int b) { return (a + b - 1) / b; }

extern "C" void kernel_launch(void* const* d_in, const int* in_sizes, int n_in,
                              void* d_out, int out_size, void* d_ws, size_t ws_size,
                              hipStream_t stream) {
    const float* u         = (const float*)d_in[0];
    const float* norm_w    = (const float*)d_in[1];
    const float* in_proj_w = (const float*)d_in[2];
    const float* conv_w    = (const float*)d_in[3];
    const float* conv_b    = (const float*)d_in[4];
    const float* x_proj_w  = (const float*)d_in[5];
    const float* dt_proj_w = (const float*)d_in[6];
    const float* dt_proj_b = (const float*)d_in[7];
    const float* A_log     = (const float*)d_in[8];
    const float* Dp        = (const float*)d_in[9];
    const float* out_proj_w= (const float*)d_in[10];
    float* out = (float*)d_out;

    // carve workspace (256B aligned chunks)
    char* p = (char*)d_ws;
    auto alloc = [&](size_t bytes) -> void* {
        void* r = (void*)p;
        p += (bytes + 255) & ~(size_t)255;
        return r;
    };
    bf16_t* unorm  = (bf16_t*)alloc((size_t)M_ * DM_ * 2);
    bf16_t* W1b    = (bf16_t*)alloc((size_t)2 * DI_ * DM_ * 2);
    bf16_t* Wxb    = (bf16_t*)alloc((size_t)XPD_ * DI_ * 2);
    bf16_t* Wdtb   = (bf16_t*)alloc((size_t)DI_ * RK_ * 2);
    bf16_t* Wob    = (bf16_t*)alloc((size_t)DM_ * DI_ * 2);
    float*  xz     = (float*) alloc((size_t)M_ * 2 * DI_ * 4);
    float*  xact   = (float*) alloc((size_t)M_ * DI_ * 4);
    bf16_t* xactb  = (bf16_t*)alloc((size_t)M_ * DI_ * 2);
    float*  xp     = (float*) alloc((size_t)M_ * XPD_ * 4);
    bf16_t* dtlowb = (bf16_t*)alloc((size_t)M_ * RK_ * 2);
    float*  dtbuf  = (float*) alloc((size_t)M_ * DI_ * 4);
    bf16_t* ybf    = (bf16_t*)alloc((size_t)M_ * DI_ * 2);

    // weight conversions f32 -> bf16
    {
        int n;
        n = 2 * DI_ * DM_; cvt_bf16_kernel<<<cdiv(n,256),256,0,stream>>>(in_proj_w, W1b, n);
        n = XPD_ * DI_;    cvt_bf16_kernel<<<cdiv(n,256),256,0,stream>>>(x_proj_w,  Wxb, n);
        n = DI_ * RK_;     cvt_bf16_kernel<<<cdiv(n,256),256,0,stream>>>(dt_proj_w, Wdtb, n);
        n = DM_ * DI_;     cvt_bf16_kernel<<<cdiv(n,256),256,0,stream>>>(out_proj_w, Wob, n);
    }

    // 1) RMSNorm -> bf16
    rmsnorm_bf16_kernel<<<M_, 256, 0, stream>>>(u, norm_w, unorm);

    // 2) in_proj GEMM: xz[M, 4096] = unorm[M,1024] * W1[4096,1024]^T
    {
        dim3 g((2 * DI_) / 64, M_ / 64);
        gemm_bf16_wmma_kernel<4, false><<<g, 128, 0, stream>>>(unorm, DM_, W1b, DM_,
                                                               xz, 2 * DI_, nullptr, DM_);
    }

    // 3) depthwise causal conv + SiLU
    conv_silu_kernel<<<(M_ * DI_) / 256, 256, 0, stream>>>(xz, conv_w, conv_b, xact, xactb);

    // 4) x_proj GEMM: xp[M, 96] = xact[M,2048] * Wx[96,2048]^T  (96 = 3 * 32 cols)
    {
        dim3 g(XPD_ / 32, M_ / 64);
        gemm_bf16_wmma_kernel<2, false><<<g, 128, 0, stream>>>(xactb, DI_, Wxb, DI_,
                                                               xp, XPD_, nullptr, DI_);
    }

    // 5) dt_low (cols 0..63 of xp) -> bf16
    {
        int n = M_ * RK_;
        cvt_bf16_strided_kernel<<<cdiv(n,256),256,0,stream>>>(xp, dtlowb, n);
    }

    // 6) dt_proj GEMM: dt_pre[M, 2048] = dt_low[M,64] * Wdt[2048,64]^T
    {
        dim3 g(DI_ / 64, M_ / 64);
        gemm_bf16_wmma_kernel<4, false><<<g, 128, 0, stream>>>(dtlowb, RK_, Wdtb, RK_,
                                                               dtbuf, DI_, nullptr, RK_);
    }

    // 7) dt = softplus(dt_pre + bias)
    dt_softplus_kernel<<<(M_ * DI_) / 256, 256, 0, stream>>>(dtbuf, dt_proj_b);

    // 8) selective scan + D skip + gating -> y (bf16)
    scan_gate_kernel<<<(B_ * DI_ * NS_) / 256, 256, 0, stream>>>(xact, dtbuf, xp, xz,
                                                                 A_log, Dp, ybf);

    // 9) out_proj GEMM + residual: out[M,1024] = y[M,2048] * Wo[1024,2048]^T + u
    {
        dim3 g(DM_ / 64, M_ / 64);
        gemm_bf16_wmma_kernel<4, true><<<g, 128, 0, stream>>>(ybf, DI_, Wob, DI_,
                                                              out, DM_, u, DI_);
    }
}
